// CausalAttention_11458972745852
// MI455X (gfx1250) — matrix operand order repaired and checked by reading, not compiled, not run on hardware
//
#include <hip/hip_runtime.h>

// Causal MHA forward for MI455X (gfx1250, wave32, WMMA, async-LDS loads).
// B=2, T=2048, C=2048, H=16, dk=128. All matmuls via v_wmma_f32_16x16x32_bf16.

#define D_MODEL 2048
#define T_SEQ   2048
#define NHEAD   16
#define DK      128
#define BATCH   2

typedef __attribute__((ext_vector_type(16))) __bf16 v16bf;
typedef __attribute__((ext_vector_type(8)))  __bf16 v8bf;
typedef __attribute__((ext_vector_type(8)))  float  v8f;

union BF16Frag { v16bf v; v8bf h[2]; };

struct __align__(16) U4 { unsigned int x, y, z, w; };

// Probe for the CDNA5 async global->LDS path (ASYNCcnt-tracked).
#if __has_builtin(__builtin_amdgcn_global_load_async_to_lds_b128) && \
    __has_builtin(__builtin_amdgcn_s_wait_asynccnt)
#define USE_ASYNC 1
#endif

#ifdef USE_ASYNC
// Exact param type from clang diagnostic: 'int __vector(4) __device__ *' (AS1)
// and by symmetry 'int __vector(4) __shared__ *' (AS3) for the LDS side.
typedef int V4I __attribute__((__vector_size__(16)));
typedef __attribute__((address_space(1))) V4I* ASYNC_GPTR;
typedef __attribute__((address_space(3))) V4I* ASYNC_LPTR;

__device__ __forceinline__ void cp16(unsigned short* ldst, const unsigned short* gsrc) {
  // flat->LDS addrspace cast == low-32-bit truncation (ISA 10.2 aperture rule)
  __builtin_amdgcn_global_load_async_to_lds_b128(
      (ASYNC_GPTR)(unsigned long long)gsrc,
      (ASYNC_LPTR)(unsigned)(unsigned long long)ldst,
      0, 0);
}
__device__ __forceinline__ void async_wait0() { __builtin_amdgcn_s_wait_asynccnt(0); }
#else
__device__ __forceinline__ void cp16(unsigned short* ldst, const unsigned short* gsrc) {
  *(U4*)ldst = *(const U4*)gsrc;
}
__device__ __forceinline__ void async_wait0() {}
#endif

__device__ __forceinline__ unsigned short f32_to_bf16(float f) {
  unsigned int u = __float_as_uint(f);
  u = u + 0x7FFFu + ((u >> 16) & 1u);   // RNE
  return (unsigned short)(u >> 16);
}

// A-matrix (16x32 bf16) fragment per ISA 7.12.2.
__device__ __forceinline__ v16bf frag_a(const unsigned short* p, int ld, int m0, int kk, int lane) {
  int m  = m0 + (lane & 15);
  int k0 = kk + ((lane >> 4) << 3);
  BF16Frag f;
  f.h[0] = *(const v8bf*)(p + m * ld + k0);
  f.h[1] = *(const v8bf*)(p + m * ld + k0 + 16);
  return f.v;
}

// B-matrix (32x16 bf16) fragment from [N][K]-contiguous storage.
__device__ __forceinline__ v16bf frag_b(const unsigned short* p, int ld, int n0, int kk, int lane) {
  int n  = n0 + (lane & 15);
  int k0 = kk + ((lane >> 4) << 3);
  BF16Frag f;
  f.h[0] = *(const v8bf*)(p + n * ld + k0);
  f.h[1] = *(const v8bf*)(p + n * ld + k0 + 16);
  return f.v;
}

__global__ void f32_to_bf16_kernel(const float* __restrict__ src,
                                   unsigned short* __restrict__ dst, int n4) {
  int i = blockIdx.x * blockDim.x + threadIdx.x;   // float4 index
  if (i < n4) {
    float4 v = ((const float4*)src)[i];
    unsigned int lo = (unsigned int)f32_to_bf16(v.x) | ((unsigned int)f32_to_bf16(v.y) << 16);
    unsigned int hi = (unsigned int)f32_to_bf16(v.z) | ((unsigned int)f32_to_bf16(v.w) << 16);
    ((unsigned int*)dst)[2 * i]     = lo;
    ((unsigned int*)dst)[2 * i + 1] = hi;
  }
}

// One 32-deep k-step: 2 A-frags + 4 B-frags -> 8 WMMA (32x64 per wave).
__device__ __forceinline__ void gemm_step(const unsigned short* As, const unsigned short* Bs,
                                          int wr, int wc, int lane, v8f acc[2][4]) {
  v16bf a0 = frag_a(As, 40, wr * 32,      0, lane);
  v16bf a1 = frag_a(As, 40, wr * 32 + 16, 0, lane);
#pragma unroll
  for (int t = 0; t < 4; ++t) {
    v16bf b = frag_b(Bs, 40, wc * 64 + t * 16, 0, lane);
    acc[0][t] = __builtin_amdgcn_wmma_f32_16x16x32_bf16(false, a0, false, b, (short)0, acc[0][t], false, false);
    acc[1][t] = __builtin_amdgcn_wmma_f32_16x16x32_bf16(false, a1, false, b, (short)0, acc[1][t], false, false);
  }
}

// C[M][N] = A[M][K] * B[N][K]^T. 128x128 tile / 256-thread block, LDS double-buffered.
__global__ __launch_bounds__(256) void gemm_bf16_wmma(
    const unsigned short* __restrict__ A,
    const unsigned short* __restrict__ B,
    void* __restrict__ C,
    int M, int N, int K, int out_bf16)
{
  __shared__ unsigned short Abuf[2][128 * 40];   // 128 rows x 32 K, padded rows (16B aligned)
  __shared__ unsigned short Bbuf[2][128 * 40];

  const int tid  = threadIdx.x;
  const int lane = tid & 31;
  const int w    = tid >> 5;
  const int row0 = blockIdx.y * 128;
  const int col0 = blockIdx.x * 128;
  const int wr   = w >> 1;   // wave row 0..3 -> rows wr*32
  const int wc   = w & 1;    // wave col 0..1 -> cols wc*64

  v8f zero = {0,0,0,0,0,0,0,0};
  v8f acc[2][4];
#pragma unroll
  for (int i = 0; i < 2; ++i)
#pragma unroll
    for (int t = 0; t < 4; ++t) acc[i][t] = zero;

  // Prologue: stage k=0 panels into buffer 0 (4x b128 per thread).
#pragma unroll
  for (int p = 0; p < 2; ++p) {
    int i = p * 256 + tid, r = i >> 2, c = (i & 3) * 8;
    cp16(&Abuf[0][r * 40 + c], &A[(size_t)(row0 + r) * K + c]);
    cp16(&Bbuf[0][r * 40 + c], &B[(size_t)(col0 + r) * K + c]);
  }
  async_wait0();
  __syncthreads();

  int cur = 0;
  for (int k0 = 0; k0 < K; k0 += 32) {
    const int nxt = cur ^ 1;
    if (k0 + 32 < K) {   // issue next panel while computing on current
#pragma unroll
      for (int p = 0; p < 2; ++p) {
        int i = p * 256 + tid, r = i >> 2, c = (i & 3) * 8;
        cp16(&Abuf[nxt][r * 40 + c], &A[(size_t)(row0 + r) * K + (k0 + 32) + c]);
        cp16(&Bbuf[nxt][r * 40 + c], &B[(size_t)(col0 + r) * K + (k0 + 32) + c]);
      }
    }
    gemm_step(Abuf[cur], Bbuf[cur], wr, wc, lane, acc);
    async_wait0();
    __syncthreads();
    cur = nxt;
  }

  const int hi8 = (lane >> 4) << 3;
#pragma unroll
  for (int rt = 0; rt < 2; ++rt)
#pragma unroll
    for (int t = 0; t < 4; ++t)
#pragma unroll
      for (int r = 0; r < 8; ++r) {
        int row = row0 + wr * 32 + rt * 16 + r + hi8;
        int col = col0 + wc * 64 + t * 16 + (lane & 15);
        if (out_bf16)
          ((unsigned short*)C)[(size_t)row * N + col] = f32_to_bf16(acc[rt][t][r]);
        else
          ((float*)C)[(size_t)row * N + col] = acc[rt][t][r];
      }
}

// Flash-style causal attention: one (b, h, 64-query tile) per 256-thread block.
__global__ __launch_bounds__(256) void attn_fwd_wmma(
    const unsigned short* __restrict__ qkv,   // [B*T][3C] bf16
    unsigned short* __restrict__ out)         // [B*T][C] bf16
{
  __shared__ unsigned short Qs[64 * 136];   // Q tile 64x128
  __shared__ unsigned short KVs[128 * 72];  // time-shared: K panel (64x136) / V^T (128x72)
  __shared__ float          Ssm[64 * 68];   // scores fp32
  __shared__ unsigned short Ps[64 * 72];    // exp(S-m) bf16
  __shared__ float mrow[64], lrow[64], arow[64];
  __shared__ float red[256];                // softmax tree reduction

  const int tid  = threadIdx.x;
  const int lane = tid & 31;
  const int w    = tid >> 5;
  const int hi8  = (lane >> 4) << 3;
  const int qt = blockIdx.x, h = blockIdx.y, b = blockIdx.z;
  const int q0   = qt * 64;
  const int bT   = b * T_SEQ;
  const int qoff = h * DK;
  const float SCALE = 0.088388347648318447f;  // 1/sqrt(128)

  // Q tile (async; completion covered by in-loop async_wait0 + barrier)
  for (int i = tid; i < 1024; i += 256) {
    int r = i >> 4, c = (i & 15) * 8;
    cp16(&Qs[r * 136 + c], &qkv[(size_t)(bT + q0 + r) * (3 * D_MODEL) + qoff + c]);
  }
  if (tid < 64) { mrow[tid] = -3.0e38f; lrow[tid] = 0.0f; }

  const int rb = w >> 1, chh = w & 1;
  const int sr = tid >> 2, sg = tid & 3;    // softmax row / 16-col segment
  v8f zero = {0,0,0,0,0,0,0,0};
  v8f oacc[4];
#pragma unroll
  for (int t = 0; t < 4; ++t) oacc[t] = zero;

  for (int j = 0; j <= qt; ++j) {
    __syncthreads();   // prior PV reads of KVs done
    for (int i = tid; i < 1024; i += 256) {   // K panel (64 x 128), ld=136
      int r = i >> 4, c = (i & 15) * 8;
      cp16(&KVs[r * 136 + c],
           &qkv[(size_t)(bT + j * 64 + r) * (3 * D_MODEL) + D_MODEL + qoff + c]);
    }
    async_wait0();
    __syncthreads();

    // S = Q K^T : two 16x16 tiles per wave, dk=128 -> 4 k-steps
    v8f s0 = zero, s1 = zero;
#pragma unroll
    for (int kk = 0; kk < 128; kk += 32) {
      v16bf a  = frag_a(Qs, 136, rb * 16, kk, lane);
      v16bf b0 = frag_b(KVs, 136, (chh * 2 + 0) * 16, kk, lane);
      v16bf b1 = frag_b(KVs, 136, (chh * 2 + 1) * 16, kk, lane);
      s0 = __builtin_amdgcn_wmma_f32_16x16x32_bf16(false, a, false, b0, (short)0, s0, false, false);
      s1 = __builtin_amdgcn_wmma_f32_16x16x32_bf16(false, a, false, b1, (short)0, s1, false, false);
    }
#pragma unroll
    for (int r = 0; r < 8; ++r) {   // scale + causal mask -> LDS
      int mm = rb * 16 + r + hi8;
      int n0 = (chh * 2 + 0) * 16 + (lane & 15);
      int n1 = (chh * 2 + 1) * 16 + (lane & 15);
      float v0 = s0[r] * SCALE;
      float v1 = s1[r] * SCALE;
      if (j * 64 + n0 > q0 + mm) v0 = -1.0e30f;
      if (j * 64 + n1 > q0 + mm) v1 = -1.0e30f;
      Ssm[mm * 68 + n0] = v0;
      Ssm[mm * 68 + n1] = v1;
    }
    __syncthreads();   // K panel free; S complete

    // V^T (128 d x 64 keys, ld=72) into KVs — overlapped with softmax phase A
    for (int i = tid; i < 1024; i += 256) {
      int r = i >> 4, c = (i & 15) * 8;
      U4 dv = *(const U4*)&qkv[(size_t)(bT + j * 64 + r) * (3 * D_MODEL) + 2 * D_MODEL + qoff + c];
      unsigned int uu[4] = {dv.x, dv.y, dv.z, dv.w};
#pragma unroll
      for (int e = 0; e < 4; ++e) {
        KVs[(c + 2 * e) * 72 + r]     = (unsigned short)(uu[e] & 0xffffu);
        KVs[(c + 2 * e + 1) * 72 + r] = (unsigned short)(uu[e] >> 16);
      }
    }
    // softmax: 4 threads per row, 16 cols each
    float pm = -3.0e38f;
    for (int c = sg * 16; c < sg * 16 + 16; ++c) pm = fmaxf(pm, Ssm[sr * 68 + c]);
    red[tid] = pm;
    __syncthreads();
    float mx = fmaxf(fmaxf(red[sr * 4 + 0], red[sr * 4 + 1]),
                     fmaxf(red[sr * 4 + 2], red[sr * 4 + 3]));
    mx = fmaxf(mx, mrow[sr]);
    __syncthreads();   // all partial maxes consumed before red[] reuse
    float ps = 0.0f;
    for (int c = sg * 16; c < sg * 16 + 16; ++c) {
      float p = __expf(Ssm[sr * 68 + c] - mx);
      Ps[sr * 72 + c] = f32_to_bf16(p);
      ps += p;
    }
    red[tid] = ps;
    __syncthreads();
    if (sg == 0) {
      float sum = red[sr * 4 + 0] + red[sr * 4 + 1] + red[sr * 4 + 2] + red[sr * 4 + 3];
      float al  = __expf(mrow[sr] - mx);
      lrow[sr] = lrow[sr] * al + sum;
      mrow[sr] = mx;
      arow[sr] = al;
    }
    __syncthreads();

    // rescale running O, then O += P * V (K=64 keys -> 2 k-steps, 4 col tiles)
#pragma unroll
    for (int t = 0; t < 4; ++t)
#pragma unroll
      for (int r = 0; r < 8; ++r)
        oacc[t][r] *= arow[rb * 16 + r + hi8];

#pragma unroll
    for (int kk = 0; kk < 64; kk += 32) {
      v16bf a = frag_a(Ps, 72, rb * 16, kk, lane);
#pragma unroll
      for (int t = 0; t < 4; ++t) {
        v16bf bv = frag_b(KVs, 72, chh * 64 + t * 16, kk, lane);
        oacc[t] = __builtin_amdgcn_wmma_f32_16x16x32_bf16(false, a, false, bv, (short)0, oacc[t], false, false);
      }
    }
  }

#pragma unroll
  for (int t = 0; t < 4; ++t)
#pragma unroll
    for (int r = 0; r < 8; ++r) {
      int row = rb * 16 + r + hi8;
      int col = chh * 64 + t * 16 + (lane & 15);
      float v = oacc[t][r] * (1.0f / lrow[row]);
      out[(size_t)(bT + q0 + row) * D_MODEL + qoff + col] = f32_to_bf16(v);
    }
}

extern "C" void kernel_launch(void* const* d_in, const int* in_sizes, int n_in,
                              void* d_out, int out_size, void* d_ws, size_t ws_size,
                              hipStream_t stream) {
  const float* x     = (const float*)d_in[0];   // [B, T, C]
  const float* Wqkv  = (const float*)d_in[1];   // [3C, C]
  const float* Wproj = (const float*)d_in[2];   // [C, C]

  const int BT = BATCH * T_SEQ;                                // 4096
  const size_t n_x   = (size_t)BT * D_MODEL;
  const size_t n_wq  = (size_t)3 * D_MODEL * D_MODEL;
  const size_t n_wp  = (size_t)D_MODEL * D_MODEL;
  const size_t n_qkv = (size_t)BT * 3 * D_MODEL;
  // workspace (u16 elems): x | Wqkv | Wproj | qkv | attn  (~112 MB)
  unsigned short* xb   = (unsigned short*)d_ws;
  unsigned short* wqb  = xb  + n_x;
  unsigned short* wpb  = wqb + n_wq;
  unsigned short* qkvb = wpb + n_wp;
  unsigned short* attb = qkvb + n_qkv;

  f32_to_bf16_kernel<<<(unsigned)(n_x  / 4 / 256), 256, 0, stream>>>(x,     xb,  (int)(n_x  / 4));
  f32_to_bf16_kernel<<<(unsigned)(n_wq / 4 / 256), 256, 0, stream>>>(Wqkv,  wqb, (int)(n_wq / 4));
  f32_to_bf16_kernel<<<(unsigned)(n_wp / 4 / 256), 256, 0, stream>>>(Wproj, wpb, (int)(n_wp / 4));

  // qkv = x @ Wqkv^T : [4096 x 6144]
  gemm_bf16_wmma<<<dim3(3 * D_MODEL / 128, BT / 128), 256, 0, stream>>>(
      xb, wqb, qkvb, BT, 3 * D_MODEL, D_MODEL, 1);

  // causal attention per (b, h, q-tile)
  attn_fwd_wmma<<<dim3(T_SEQ / 64, NHEAD, BATCH), 256, 0, stream>>>(qkvb, attb);

  // out = attn @ Wproj^T : [4096 x 2048] fp32
  gemm_bf16_wmma<<<dim3(D_MODEL / 128, BT / 128), 256, 0, stream>>>(
      attb, wpb, d_out, BT, D_MODEL, D_MODEL, 0);
}
